// Model_35931696398572
// MI455X (gfx1250) — compile-verified, matrix operand order checked by reading, not run on hardware
//
#include <hip/hip_runtime.h>

// ---------------------------------------------------------------------------
// DGCNN-ish model for MI455X (gfx1250, wave32, WMMA).
// Heavy contractions (pairwise inner products for knn #2, block6/7/8 MLPs)
// run on v_wmma_f32_16x16x32_f16 (f16 in, f32 accumulate). Irregular work
// (top-k, gathers, 6-wide edge conv) stays in fp32 VALU.
// ---------------------------------------------------------------------------

typedef __attribute__((ext_vector_type(16))) _Float16 v16h;
typedef __attribute__((ext_vector_type(8)))  _Float16 v8h;
typedef __attribute__((ext_vector_type(8)))  float    v8f;

#define BN_EPS 1e-5f
#define NEG_INF (-3.402823466e38f)

static constexpr int BATCH = 4;
static constexpr int NPTS  = 4096;
static constexpr int KNN   = 10;

// ---- workspace layout (bytes, all 256-aligned) ----------------------------
static constexpr size_t OFF_IDX1 = 0;                       // B*N*K int      = 655360
static constexpr size_t OFF_IDX2 = 655360;                  // B*N*K int      = 655360
static constexpr size_t OFF_XX2  = 1310720;                 // B*N f32        = 65536
static constexpr size_t OFF_CAT1 = 1376256;                 // B*N*128 f16    = 4194304  [x1|x2]
static constexpr size_t OFF_GMAX = 5570560;                 // B*1024 f32     = 16384
static constexpr size_t OFF_W6H  = 5586944;                 // 1024*128 f16   = 262144
static constexpr size_t OFF_W7H  = 5849088;                 // 512*1152 f16   = 1179648
static constexpr size_t OFF_W8H  = 7028736;                 // 256*512 f16    = 262144
static constexpr size_t OFF_CAT2 = 7290880;                 // B*N*1152 f16   = 37748736
static constexpr size_t OFF_H7H  = 45039616;                // B*N*512 f16    = 16777216
static constexpr size_t OFF_H8H  = 61816832;                // B*N*256 f16    = 8388608
// total ~70.2 MB

// ---------------------------------------------------------------------------
__device__ __forceinline__ float mishf(float y) {
    float sp = (y > 20.0f) ? y : log1pf(expf(y));
    return y * tanhf(sp);
}

// monotonic-bits float atomic max (lowers to global_atomic_max/min)
__device__ __forceinline__ void atomicMaxF(float* addr, float val) {
    if (val >= 0.0f) atomicMax((int*)addr, __float_as_int(val));
    else             atomicMin((unsigned int*)addr, __float_as_uint(val));
}

// 16-bit A/B fragment per CDNA5 ISA layout: lane holds 8 contiguous halves at
// kBase and 8 at kBase+16 (kBase = 0 for lanes 0-15, 8 for lanes 16-31).
__device__ __forceinline__ v16h load_frag(const _Float16* p) {
    v8h lo = *(const v8h*)(p);
    v8h hi = *(const v8h*)(p + 16);
    return __builtin_shufflevector(lo, hi, 0,1,2,3,4,5,6,7,8,9,10,11,12,13,14,15);
}

__device__ __forceinline__ v8f wmma_f16(v16h a, v16h b, v8f c) {
    return __builtin_amdgcn_wmma_f32_16x16x32_f16(false, a, false, b, (short)0, c,
                                                  false, false);
}

// ---------------------------------------------------------------------------
__global__ void cvt_f16_kernel(const float* __restrict__ s, _Float16* __restrict__ d, int n) {
    int i = blockIdx.x * 256 + threadIdx.x;
    if (i < n) d[i] = (_Float16)s[i];
}

__global__ void fill_neginf_kernel(float* __restrict__ p, int n) {
    int i = blockIdx.x * 256 + threadIdx.x;
    if (i < n) p[i] = NEG_INF;
}

// ---------------------------------------------------------------------------
// knn on raw xyz: neg_dist = -||p_n - p_m||^2 ; full cloud staged in LDS.
__global__ void knn1_kernel(const float* __restrict__ x, int* __restrict__ idx1) {
    __shared__ float sx[NPTS], sy[NPTS], sz[NPTS];
    const int tid = threadIdx.x;
    const int blocksPerB = NPTS / 256;
    const int b  = blockIdx.x / blocksPerB;
    const int nb = blockIdx.x % blocksPerB;
    const float* xb = x + (size_t)b * 3 * NPTS;
    for (int i = tid; i < NPTS; i += 256) {
        sx[i] = xb[i]; sy[i] = xb[NPTS + i]; sz[i] = xb[2 * NPTS + i];
    }
    __syncthreads();
    const int n = nb * 256 + tid;
    const float cx = sx[n], cy = sy[n], cz = sz[n];
    float vals[KNN]; int idxs[KNN];
#pragma unroll
    for (int p = 0; p < KNN; ++p) { vals[p] = NEG_INF; idxs[p] = 0; }
    for (int m = 0; m < NPTS; ++m) {
        float dx = sx[m] - cx, dy = sy[m] - cy, dz = sz[m] - cz;
        float v = -(dx * dx + dy * dy + dz * dz);
        if (v > vals[KNN - 1]) {
            vals[KNN - 1] = v; idxs[KNN - 1] = m;
#pragma unroll
            for (int p = KNN - 1; p > 0; --p) {
                if (vals[p] > vals[p - 1]) {
                    float tv = vals[p]; vals[p] = vals[p - 1]; vals[p - 1] = tv;
                    int   ti = idxs[p]; idxs[p] = idxs[p - 1]; idxs[p - 1] = ti;
                }
            }
        }
    }
    int* op = idx1 + ((size_t)b * NPTS + n) * KNN;
#pragma unroll
    for (int p = 0; p < KNN; ++p) op[p] = idxs[p];
}

// ---------------------------------------------------------------------------
// Edge conv stage 1: (6 -> 64 -> 64), max over k. One wave per point,
// 2 output channels per lane, h1 shared via shfl. Writes x1 (f16) + |x1|^2.
__global__ void __launch_bounds__(32)
edgeconv1_kernel(const float* __restrict__ x, const int* __restrict__ idx1,
                 const float* __restrict__ w1, const float* __restrict__ g1,
                 const float* __restrict__ b1, const float* __restrict__ m1,
                 const float* __restrict__ v1,
                 const float* __restrict__ w2, const float* __restrict__ g2,
                 const float* __restrict__ b2, const float* __restrict__ m2,
                 const float* __restrict__ v2,
                 _Float16* __restrict__ cat1h, float* __restrict__ xx2) {
    const int p = blockIdx.x;
    const int b = p >> 12, n = p & (NPTS - 1);
    const int lane = threadIdx.x;
    const int c0 = lane, c1 = lane + 32;
    const float* xb = x + (size_t)b * 3 * NPTS;
    const float cx = xb[n], cy = xb[NPTS + n], cz = xb[2 * NPTS + n];

    const float s1a = g1[c0] * rsqrtf(v1[c0] + BN_EPS), t1a = b1[c0] - m1[c0] * s1a;
    const float s1b = g1[c1] * rsqrtf(v1[c1] + BN_EPS), t1b = b1[c1] - m1[c1] * s1b;
    const float s2a = g2[c0] * rsqrtf(v2[c0] + BN_EPS), t2a = b2[c0] - m2[c0] * s2a;
    const float s2b = g2[c1] * rsqrtf(v2[c1] + BN_EPS), t2b = b2[c1] - m2[c1] * s2b;

    const float wa0 = w1[c0*6+0], wa1 = w1[c0*6+1], wa2 = w1[c0*6+2];
    const float wa3 = w1[c0*6+3], wa4 = w1[c0*6+4], wa5 = w1[c0*6+5];
    const float wb0 = w1[c1*6+0], wb1 = w1[c1*6+1], wb2 = w1[c1*6+2];
    const float wb3 = w1[c1*6+3], wb4 = w1[c1*6+4], wb5 = w1[c1*6+5];

    const int* ip = idx1 + (size_t)p * KNN;
    float h0[KNN], h1v[KNN];
#pragma unroll
    for (int kk = 0; kk < KNN; ++kk) {
        int m = ip[kk];
        float e0 = xb[m] - cx, e1 = xb[NPTS + m] - cy, e2 = xb[2 * NPTS + m] - cz;
        float q0 = wa0*e0 + wa1*e1 + wa2*e2 + wa3*cx + wa4*cy + wa5*cz;
        float q1 = wb0*e0 + wb1*e1 + wb2*e2 + wb3*cx + wb4*cy + wb5*cz;
        h0[kk]  = mishf(q0 * s1a + t1a);
        h1v[kk] = mishf(q1 * s1b + t1b);
    }
    float s0[KNN], s1[KNN];
#pragma unroll
    for (int kk = 0; kk < KNN; ++kk) { s0[kk] = 0.f; s1[kk] = 0.f; }
    for (int j = 0; j < 32; ++j) {
        float w2a0 = w2[c0*64 + j], w2a1 = w2[c0*64 + 32 + j];
        float w2b0 = w2[c1*64 + j], w2b1 = w2[c1*64 + 32 + j];
#pragma unroll
        for (int kk = 0; kk < KNN; ++kk) {
            float a  = __shfl(h0[kk],  j, 32);
            float bq = __shfl(h1v[kk], j, 32);
            s0[kk] += w2a0 * a + w2a1 * bq;
            s1[kk] += w2b0 * a + w2b1 * bq;
        }
    }
    float acc0 = NEG_INF, acc1 = NEG_INF;
#pragma unroll
    for (int kk = 0; kk < KNN; ++kk) {
        acc0 = fmaxf(acc0, mishf(s0[kk] * s2a + t2a));
        acc1 = fmaxf(acc1, mishf(s1[kk] * s2b + t2b));
    }
    cat1h[(size_t)p * 128 + c0] = (_Float16)acc0;
    cat1h[(size_t)p * 128 + c1] = (_Float16)acc1;
    float ss = acc0 * acc0 + acc1 * acc1;
    for (int o = 16; o > 0; o >>= 1) ss += __shfl_xor(ss, o, 32);
    if (lane == 0) xx2[p] = ss;
}

// ---------------------------------------------------------------------------
// knn on 64-d features: WMMA inner products. Each wave owns 16 rows; per
// iteration it computes a 16x64 neg-dist tile (8 chained 16x16x32 WMMAs),
// spills it to LDS, and 16 lanes run the top-k insertion scan. 64 columns
// per barrier pair amortizes sync overhead 4x vs a 16-wide tile.
__global__ void __launch_bounds__(128)
knn2_kernel(const _Float16* __restrict__ cat1h, const float* __restrict__ xx2,
            int* __restrict__ idx2) {
    __shared__ float tile[4][16][65];
    const int lane = threadIdx.x & 31;
    const int wave = threadIdx.x >> 5;
    const int grp  = blockIdx.x;                       // 64 rows / block
    const int b    = (grp * 64) >> 12;
    const int nBase = ((grp * 64) & (NPTS - 1)) + wave * 16;
    const _Float16* feat = cat1h + (size_t)b * NPTS * 128;
    const float*    xxb  = xx2 + (size_t)b * NPTS;
    const int laneRow = nBase + (lane & 15);
    const int kSel    = (lane < 16) ? 0 : 8;
    const int rowHi   = (lane >= 16) ? 8 : 0;

    v16h a0 = load_frag(feat + (size_t)laneRow * 128 + kSel);
    v16h a1 = load_frag(feat + (size_t)laneRow * 128 + 32 + kSel);
    float xxR[8];
#pragma unroll
    for (int r = 0; r < 8; ++r) xxR[r] = xxb[nBase + r + rowHi];

    float vals[KNN]; int idxs[KNN];
#pragma unroll
    for (int p = 0; p < KNN; ++p) { vals[p] = NEG_INF; idxs[p] = 0; }

    for (int mBase = 0; mBase < NPTS; mBase += 64) {
#pragma unroll
        for (int t = 0; t < 4; ++t) {
            const int colPt = mBase + t * 16 + (lane & 15);
            v16h b0 = load_frag(feat + (size_t)colPt * 128 + kSel);
            v16h b1 = load_frag(feat + (size_t)colPt * 128 + 32 + kSel);
            v8f c = {};
            c = wmma_f16(a0, b0, c);
            c = wmma_f16(a1, b1, c);
            const float xxC = xxb[colPt];
#pragma unroll
            for (int r = 0; r < 8; ++r)
                tile[wave][r + rowHi][t * 16 + (lane & 15)] =
                    2.0f * c[r] - xxR[r] - xxC;
        }
        __syncthreads();
        if (lane < 16) {
            for (int j = 0; j < 64; ++j) {
                float v = tile[wave][lane][j];
                int   m = mBase + j;
                if (v > vals[KNN - 1]) {
                    vals[KNN - 1] = v; idxs[KNN - 1] = m;
#pragma unroll
                    for (int q = KNN - 1; q > 0; --q) {
                        if (vals[q] > vals[q - 1]) {
                            float tv = vals[q]; vals[q] = vals[q - 1]; vals[q - 1] = tv;
                            int   ti = idxs[q]; idxs[q] = idxs[q - 1]; idxs[q - 1] = ti;
                        }
                    }
                }
            }
        }
        __syncthreads();
    }
    if (lane < 16) {
        int* op = idx2 + ((size_t)b * NPTS + nBase + lane) * KNN;
#pragma unroll
        for (int p = 0; p < KNN; ++p) op[p] = idxs[p];
    }
}

// ---------------------------------------------------------------------------
// Edge conv stage 2: (128 -> 64 -> 64), max over k. Reads x1 half of cat1h,
// writes x2 half (disjoint addresses).
__global__ void __launch_bounds__(32)
edgeconv2_kernel(const _Float16* __restrict__ cat1h, const int* __restrict__ idx2,
                 const float* __restrict__ w3, const float* __restrict__ g3,
                 const float* __restrict__ b3, const float* __restrict__ m3,
                 const float* __restrict__ v3,
                 const float* __restrict__ w4, const float* __restrict__ g4,
                 const float* __restrict__ b4, const float* __restrict__ m4,
                 const float* __restrict__ v4,
                 _Float16* __restrict__ cat1h_out) {
    const int p = blockIdx.x;
    const int b = p >> 12, n = p & (NPTS - 1);
    const int lane = threadIdx.x;
    const int c0 = lane, c1 = lane + 32;
    const _Float16* fb = cat1h + (size_t)b * NPTS * 128;

    const float s3a = g3[c0] * rsqrtf(v3[c0] + BN_EPS), t3a = b3[c0] - m3[c0] * s3a;
    const float s3b = g3[c1] * rsqrtf(v3[c1] + BN_EPS), t3b = b3[c1] - m3[c1] * s3b;
    const float s4a = g4[c0] * rsqrtf(v4[c0] + BN_EPS), t4a = b4[c0] - m4[c0] * s4a;
    const float s4b = g4[c1] * rsqrtf(v4[c1] + BN_EPS), t4b = b4[c1] - m4[c1] * s4b;

    const float ct0 = (float)fb[(size_t)n * 128 + lane];
    const float ct1 = (float)fb[(size_t)n * 128 + 32 + lane];

    const int* ip = idx2 + (size_t)p * KNN;
    float nb0[KNN], nb1[KNN];
#pragma unroll
    for (int kk = 0; kk < KNN; ++kk) {
        int m = ip[kk];
        nb0[kk] = (float)fb[(size_t)m * 128 + lane];
        nb1[kk] = (float)fb[(size_t)m * 128 + 32 + lane];
    }
    float s0[KNN], s1[KNN];
#pragma unroll
    for (int kk = 0; kk < KNN; ++kk) { s0[kk] = 0.f; s1[kk] = 0.f; }
    for (int j = 0; j < 32; ++j) {
        // edge dims d=j (lane j of ct0/nb0) and d=32+j (lane j of ct1/nb1)
        float w3a0 = w3[c0*128 + j],      w3a1 = w3[c0*128 + 64 + j];
        float w3a2 = w3[c0*128 + 32 + j], w3a3 = w3[c0*128 + 96 + j];
        float w3b0 = w3[c1*128 + j],      w3b1 = w3[c1*128 + 64 + j];
        float w3b2 = w3[c1*128 + 32 + j], w3b3 = w3[c1*128 + 96 + j];
        float cA = __shfl(ct0, j, 32), cB = __shfl(ct1, j, 32);
        float base0 = (w3a1 - w3a0) * cA + (w3a3 - w3a2) * cB;
        float base1 = (w3b1 - w3b0) * cA + (w3b3 - w3b2) * cB;
#pragma unroll
        for (int kk = 0; kk < KNN; ++kk) {
            float nA = __shfl(nb0[kk], j, 32);
            float nB = __shfl(nb1[kk], j, 32);
            s0[kk] += w3a0 * nA + w3a2 * nB + base0;
            s1[kk] += w3b0 * nA + w3b2 * nB + base1;
        }
    }
    float h0[KNN], h1v[KNN];
#pragma unroll
    for (int kk = 0; kk < KNN; ++kk) {
        h0[kk]  = mishf(s0[kk] * s3a + t3a);
        h1v[kk] = mishf(s1[kk] * s3b + t3b);
        s0[kk] = 0.f; s1[kk] = 0.f;
    }
    for (int j = 0; j < 32; ++j) {
        float w4a0 = w4[c0*64 + j], w4a1 = w4[c0*64 + 32 + j];
        float w4b0 = w4[c1*64 + j], w4b1 = w4[c1*64 + 32 + j];
#pragma unroll
        for (int kk = 0; kk < KNN; ++kk) {
            float a  = __shfl(h0[kk],  j, 32);
            float bq = __shfl(h1v[kk], j, 32);
            s0[kk] += w4a0 * a + w4a1 * bq;
            s1[kk] += w4b0 * a + w4b1 * bq;
        }
    }
    float acc0 = NEG_INF, acc1 = NEG_INF;
#pragma unroll
    for (int kk = 0; kk < KNN; ++kk) {
        acc0 = fmaxf(acc0, mishf(s0[kk] * s4a + t4a));
        acc1 = fmaxf(acc1, mishf(s1[kk] * s4b + t4b));
    }
    cat1h_out[(size_t)p * 128 + 64 + c0] = (_Float16)acc0;
    cat1h_out[(size_t)p * 128 + 64 + c1] = (_Float16)acc1;
}

// ---------------------------------------------------------------------------
// Generic WMMA GEMM: rows = points, cols = output channels, K = KCHUNKS*32.
// Wave computes 16 rows x 64 cols (4 accumulator tiles). Epilogue = BN+Mish,
// then either f16 store or atomic row-max reduction into gmax (block6).
template<int KCHUNKS, bool ATOMICMAX>
__global__ void __launch_bounds__(128)
gemm_bn_mish(const _Float16* __restrict__ A, int lda,
             const _Float16* __restrict__ W,
             const float* __restrict__ gg, const float* __restrict__ bb,
             const float* __restrict__ mm, const float* __restrict__ vv,
             _Float16* __restrict__ out, int ldo,
             float* __restrict__ gmax, int gcols) {
    const int lane    = threadIdx.x & 31;
    const int wave    = threadIdx.x >> 5;
    const int rowBase = blockIdx.x * 16;
    const int colBase = blockIdx.y * 256 + wave * 64;
    const int laneRow = rowBase + (lane & 15);
    const int laneCol = lane & 15;
    const int kSel    = (lane < 16) ? 0 : 8;
    const int rowHi   = (lane >= 16) ? 8 : 0;

    v8f acc[4] = {v8f{}, v8f{}, v8f{}, v8f{}};
    const _Float16* aRow = A + (size_t)laneRow * lda + kSel;
#pragma unroll 4
    for (int ck = 0; ck < KCHUNKS; ++ck) {
        v16h a = load_frag(aRow + ck * 32);
#pragma unroll
        for (int t = 0; t < 4; ++t) {
            const _Float16* wp = W + (size_t)(colBase + t * 16 + laneCol) * (KCHUNKS * 32)
                                   + ck * 32 + kSel;
            acc[t] = wmma_f16(a, load_frag(wp), acc[t]);
        }
    }
#pragma unroll
    for (int t = 0; t < 4; ++t) {
        const int col = colBase + t * 16 + laneCol;
        const float sc = gg[col] * rsqrtf(vv[col] + BN_EPS);
        const float sh = bb[col] - mm[col] * sc;
        if constexpr (ATOMICMAX) {
            float lm = NEG_INF;
#pragma unroll
            for (int r = 0; r < 8; ++r) lm = fmaxf(lm, mishf(acc[t][r] * sc + sh));
            const int bidx = rowBase >> 12;   // N = 4096 rows per batch
            atomicMaxF(&gmax[(size_t)bidx * gcols + col], lm);
        } else {
#pragma unroll
            for (int r = 0; r < 8; ++r) {
                const int row = rowBase + r + rowHi;
                out[(size_t)row * ldo + col] = (_Float16)mishf(acc[t][r] * sc + sh);
            }
        }
    }
}

// ---------------------------------------------------------------------------
__global__ void cat2_prep_kernel(const float* __restrict__ gmax,
                                 const _Float16* __restrict__ cat1h,
                                 _Float16* __restrict__ cat2h) {
    size_t i  = (size_t)blockIdx.x * 256 + threadIdx.x;   // B*N*1152 total
    size_t bn = i / 1152;
    int    k  = (int)(i % 1152);
    int    b  = (int)(bn >> 12);
    _Float16 v;
    if (k < 1024) v = (_Float16)gmax[(size_t)b * 1024 + k];
    else          v = cat1h[bn * 128 + (k - 1024)];   // [x1|x2] matches cat order
    cat2h[i] = v;
}

__global__ void final_kernel(const _Float16* __restrict__ h8h,
                             const float* __restrict__ w9,
                             float* __restrict__ out) {
    const int p = blockIdx.x * 256 + threadIdx.x;   // B*N
    const int b = p >> 12, n = p & (NPTS - 1);
    const _Float16* hp = h8h + (size_t)p * 256;
    float s0 = 0.f, s1 = 0.f;
#pragma unroll 4
    for (int c = 0; c < 256; ++c) {
        float h = (float)hp[c];
        s0 += w9[c] * h;
        s1 += w9[256 + c] * h;
    }
    out[(size_t)b * 2 * NPTS + n]        = s0;
    out[(size_t)b * 2 * NPTS + NPTS + n] = s1;
}

// ---------------------------------------------------------------------------
extern "C" void kernel_launch(void* const* d_in, const int* in_sizes, int n_in,
                              void* d_out, int out_size, void* d_ws, size_t ws_size,
                              hipStream_t stream) {
    const float* x  = (const float*)d_in[0];
    const float* w1 = (const float*)d_in[1];
    const float *g1 = (const float*)d_in[2], *b1 = (const float*)d_in[3],
                *m1 = (const float*)d_in[4], *v1 = (const float*)d_in[5];
    const float* w2 = (const float*)d_in[6];
    const float *g2 = (const float*)d_in[7], *b2 = (const float*)d_in[8],
                *m2 = (const float*)d_in[9], *v2 = (const float*)d_in[10];
    const float* w3 = (const float*)d_in[11];
    const float *g3 = (const float*)d_in[12], *b3 = (const float*)d_in[13],
                *m3 = (const float*)d_in[14], *v3 = (const float*)d_in[15];
    const float* w4 = (const float*)d_in[16];
    const float *g4 = (const float*)d_in[17], *b4 = (const float*)d_in[18],
                *m4 = (const float*)d_in[19], *v4 = (const float*)d_in[20];
    const float* w6 = (const float*)d_in[21];
    const float *g6 = (const float*)d_in[22], *b6 = (const float*)d_in[23],
                *m6 = (const float*)d_in[24], *v6 = (const float*)d_in[25];
    const float* w7 = (const float*)d_in[26];
    const float *g7 = (const float*)d_in[27], *b7 = (const float*)d_in[28],
                *m7 = (const float*)d_in[29], *v7 = (const float*)d_in[30];
    const float* w8 = (const float*)d_in[31];
    const float *g8 = (const float*)d_in[32], *b8 = (const float*)d_in[33],
                *m8 = (const float*)d_in[34], *v8 = (const float*)d_in[35];
    const float* w9 = (const float*)d_in[36];

    char* ws = (char*)d_ws;
    int*       idx1  = (int*)      (ws + OFF_IDX1);
    int*       idx2  = (int*)      (ws + OFF_IDX2);
    float*     xx2   = (float*)    (ws + OFF_XX2);
    _Float16*  cat1h = (_Float16*) (ws + OFF_CAT1);
    float*     gmax  = (float*)    (ws + OFF_GMAX);
    _Float16*  w6h   = (_Float16*) (ws + OFF_W6H);
    _Float16*  w7h   = (_Float16*) (ws + OFF_W7H);
    _Float16*  w8h   = (_Float16*) (ws + OFF_W8H);
    _Float16*  cat2h = (_Float16*) (ws + OFF_CAT2);
    _Float16*  h7h   = (_Float16*) (ws + OFF_H7H);
    _Float16*  h8h   = (_Float16*) (ws + OFF_H8H);

    // weight conversion + gmax init
    cvt_f16_kernel<<<(1024*128 + 255) / 256, 256, 0, stream>>>(w6, w6h, 1024*128);
    cvt_f16_kernel<<<(512*1152 + 255) / 256, 256, 0, stream>>>(w7, w7h, 512*1152);
    cvt_f16_kernel<<<(256*512  + 255) / 256, 256, 0, stream>>>(w8, w8h, 256*512);
    fill_neginf_kernel<<<(BATCH*1024 + 255) / 256, 256, 0, stream>>>(gmax, BATCH*1024);

    // graph feature 1 + edge conv 1
    knn1_kernel<<<BATCH * (NPTS / 256), 256, 0, stream>>>(x, idx1);
    edgeconv1_kernel<<<BATCH * NPTS, 32, 0, stream>>>(x, idx1,
        w1, g1, b1, m1, v1, w2, g2, b2, m2, v2, cat1h, xx2);

    // graph feature 2 (WMMA inner products) + edge conv 2
    knn2_kernel<<<BATCH * NPTS / 64, 128, 0, stream>>>(cat1h, xx2, idx2);
    edgeconv2_kernel<<<BATCH * NPTS, 32, 0, stream>>>(cat1h, idx2,
        w3, g3, b3, m3, v3, w4, g4, b4, m4, v4, cat1h);

    // block6 (128->1024) fused with global max reduction
    gemm_bn_mish<4, true><<<dim3(BATCH * NPTS / 16, 4), 128, 0, stream>>>(
        cat1h, 128, w6h, g6, b6, m6, v6, nullptr, 0, gmax, 1024);

    // cat2 = [gmax | x1 | x2]
    cat2_prep_kernel<<<(BATCH * NPTS * 1152) / 256, 256, 0, stream>>>(gmax, cat1h, cat2h);

    // block7 (1152->512), block8 (512->256)
    gemm_bn_mish<36, false><<<dim3(BATCH * NPTS / 16, 2), 128, 0, stream>>>(
        cat2h, 1152, w7h, g7, b7, m7, v7, h7h, 512, nullptr, 0);
    gemm_bn_mish<16, false><<<dim3(BATCH * NPTS / 16, 1), 128, 0, stream>>>(
        h7h, 512, w8h, g8, b8, m8, v8, h8h, 256, nullptr, 0);

    // final 256 -> 2 projection, out [B,2,N] f32
    final_kernel<<<BATCH * NPTS / 256, 256, 0, stream>>>(h8h, w9, (float*)d_out);
}
